// Fusion_3161095930117
// MI455X (gfx1250) — matrix operand order repaired and checked by reading, not compiled, not run on hardware
//
#include <hip/hip_runtime.h>
#include <hip/hip_bf16.h>

// ---------------- problem constants (baked from reference) ----------------
#define D_MODEL  256
#define NHEADS   8
#define NLEVELS  4
#define NPOINTS  20
#define DHEAD    32
#define NBATCH   4
#define LQ       2048
#define LENIN    19560

#define BROW 132          // padded f32 row stride of the LDS B tile (128 + 4)

typedef __attribute__((ext_vector_type(16))) __bf16 v16bf;
typedef __attribute__((ext_vector_type(8)))  float  v8f;

union FragBF { unsigned int u[8]; v16bf v; };

// f32 pair -> packed bf16 (round-to-nearest) in one v_perm_b32
__device__ __forceinline__ unsigned int f2bf2(float lo, float hi) {
    unsigned int a = __float_as_uint(lo) + 0x8000u;
    unsigned int b = __float_as_uint(hi) + 0x8000u;
    // result bytes: [1:0] = a[3:2], [3:2] = b[3:2]  (sel: s1 bytes 0-3, s0 bytes 4-7)
    return __builtin_amdgcn_perm(b, a, 0x07060302u);
}

__device__ __forceinline__ unsigned short f2bf1(float f) {
    return (unsigned short)((__float_as_uint(f) + 0x8000u) >> 16);
}
__device__ __forceinline__ float bf2f(unsigned short u) {
    return __uint_as_float((unsigned int)u << 16);
}

// Async-DMA one 32x128 f32 B tile (rows k0..k0+31, cols n0..n0+127) into LDS.
// 64 threads; thread t owns vec4 columns (t&31)*4 of rows (t>>5)+2r.
// Each issue: GLOBAL_LOAD_ASYNC_TO_LDS_B128 (16B per lane, ASYNCcnt-tracked,
// no VGPR staging). 16 issues per wave per tile.
__device__ __forceinline__ void stage_b_async(const float* __restrict__ B,
                                              int Ncols, int n0, int k0, int t,
                                              unsigned ldsBase) {
    unsigned lds = ldsBase + (unsigned)(((t >> 5) * BROW + (t & 31) * 4) * 4);
    const float* g = B + (size_t)(k0 + (t >> 5)) * Ncols + n0 + (t & 31) * 4;
    #pragma unroll 1
    for (int r = 0; r < 16; ++r) {
        asm volatile("global_load_async_to_lds_b128 %0, %1, off"
                     :: "v"(lds), "v"(g) : "memory");
        lds += 2u * BROW * 4u;
        g   += 2 * (size_t)Ncols;
    }
}

// ---------------------------------------------------------------------------
// Tiled GEMM: Out[M,Ncols] = A[M,256] @ B[256,Ncols] + bias
// 2 waves / block; block tile 32(M) x 128(N); each wave computes a 16x128
// strip as 8 accumulators sharing one A fragment (8 wmma per A-frag load).
// B tiles move global->LDS via async-B128 DMA, double-buffered: while tile i
// is consumed, tile i+1 is in flight; s_wait_asynccnt 16 releases tile i
// (async completions are in-order). A tile converted to bf16 in VGPRs (small).
// mode==1: scatter output as round-to-nearest (ties away) bf16 into value layout [n][h][pos][32]
//          (halves the gather traffic of the downstream sampler).
// ---------------------------------------------------------------------------
__global__ __launch_bounds__(64) void gemm_bf16_wmma(
    const float* __restrict__ A, const float* __restrict__ B,
    const float* __restrict__ bias, void* __restrict__ OutRaw,
    int M, int Ncols, int mode)
{
    __shared__ unsigned int sA[32 * 17];   // bf16 pairs [m][kpair], stride 17 u32
    __shared__ float sBf[2][32 * BROW];    // f32 ping-pong B tiles [k][n]

    const int t    = threadIdx.x;
    const int lane = t & 31;
    const int w    = t >> 5;
    const int m0   = blockIdx.x * 32;
    const int n0   = blockIdx.y * 128;

    v8f acc[8];
    #pragma unroll
    for (int j = 0; j < 8; ++j) acc[j] = v8f{0.f,0.f,0.f,0.f,0.f,0.f,0.f,0.f};

    // preamble: async-stage B tile 0 into buffer 0
    stage_b_async(B, Ncols, n0, 0, t, (unsigned)(size_t)&sBf[0][0]);

    #pragma unroll 1
    for (int i = 0; i < D_MODEL / 32; ++i) {
        const int k0 = i * 32;
        // stage A tile 32x32 -> bf16 (each thread: one half-row of 16 f32)
        {
            const int row = t >> 1;
            const int c0  = (t & 1) * 16;
            const float* ap = A + (size_t)(m0 + row) * D_MODEL + k0 + c0;
            #pragma unroll
            for (int j = 0; j < 16; j += 2)
                sA[row * 17 + ((c0 + j) >> 1)] = f2bf2(ap[j], ap[j + 1]);
        }
        // kick off next B tile, then wait for the current one (in-order:
        // <=16 outstanding means tile i has fully landed in LDS)
        if (i + 1 < D_MODEL / 32) {
            stage_b_async(B, Ncols, n0, k0 + 32, t,
                          (unsigned)(size_t)&sBf[(i + 1) & 1][0]);
            asm volatile("s_wait_asynccnt 0x10" ::: "memory");
        } else {
            asm volatile("s_wait_asynccnt 0x0" ::: "memory");
        }
        __syncthreads();

        // fragments per ISA 7.12.2 lane layouts
        const int m  = lane & 15;
        const int nn = lane & 15;
        const int kh = lane >> 4;
        const float* bf = &sBf[i & 1][0];

        FragBF a;
        #pragma unroll
        for (int v = 0; v < 8; ++v) {
            // A: VGPR v holds K-pair {kbA,kbA+1}; lane halves interleave K by 8
            const int kbA = ((v < 4) ? 2 * v : 16 + 2 * (v - 4)) + kh * 8;
            a.u[v] = sA[(16 * w + m) * 17 + (kbA >> 1)];
        }
        #pragma unroll
        for (int j = 0; j < 8; ++j) {
            FragBF b;
            #pragma unroll
            for (int v = 0; v < 8; ++v) {
                // B: VGPR v holds K-pair {kb,kb+1}; lane halves interleave K by 8 (matches A)
                const int kb  = ((v < 4) ? 2 * v : 16 + 2 * (v - 4)) + kh * 8;
                const int col = j * 16 + nn;
                b.u[v] = f2bf2(bf[kb * BROW + col], bf[(kb + 1) * BROW + col]);
            }
            acc[j] = __builtin_amdgcn_wmma_f32_16x16x32_bf16(
                         false, a.v, false, b.v, (short)0, acc[j], false, false);
        }
        __syncthreads();
    }

    // epilogue: D layout -> lane holds col (lane&15); VGPR r -> row +r / +8 half
    #pragma unroll
    for (int j = 0; j < 8; ++j) {
        const int col = n0 + j * 16 + (lane & 15);
        const float bv = bias[col];
        #pragma unroll
        for (int r = 0; r < 8; ++r) {
            const int row = m0 + 16 * w + r + 8 * (lane >> 4);
            if (row < M) {
                const float val = acc[j][r] + bv;
                if (mode == 1) {
                    const int n   = row / LENIN;
                    const int pos = row - n * LENIN;
                    const int h   = col >> 5;
                    const int d   = col & 31;
                    ((unsigned short*)OutRaw)[
                        (((size_t)(n * NHEADS + h) * LENIN) + pos) * DHEAD + d] =
                        f2bf1(val);
                } else {
                    ((float*)OutRaw)[(size_t)row * Ncols + col] = val;
                }
            }
        }
    }
}

// ---------------------------------------------------------------------------
// Softmax over 80 logits per (n,q,h); one wave per row, in-place.
// ---------------------------------------------------------------------------
__global__ __launch_bounds__(128) void softmax80(float* __restrict__ logits,
                                                 int rows)
{
    const int row  = blockIdx.x * 4 + (threadIdx.x >> 5);
    const int lane = threadIdx.x & 31;
    if (row >= rows) return;
    float* lp = logits + (size_t)row * (NLEVELS * NPOINTS);

    float v0 = lp[lane];
    float v1 = lp[lane + 32];
    float v2 = (lane < 16) ? lp[lane + 64] : -3.0e38f;

    float m = fmaxf(v0, fmaxf(v1, v2));
    #pragma unroll
    for (int o = 16; o > 0; o >>= 1) m = fmaxf(m, __shfl_xor(m, o, 32));

    float e0 = __expf(v0 - m);
    float e1 = __expf(v1 - m);
    float e2 = (lane < 16) ? __expf(v2 - m) : 0.f;
    float s  = e0 + e1 + e2;
    #pragma unroll
    for (int o = 16; o > 0; o >>= 1) s += __shfl_xor(s, o, 32);

    const float inv = 1.0f / s;
    lp[lane]      = e0 * inv;
    lp[lane + 32] = e1 * inv;
    if (lane < 16) lp[lane + 64] = e2 * inv;
}

// ---------------------------------------------------------------------------
// Bilinear sampling + attention-weighted accumulation.
// One wave per (n,q,h); 32 lanes = 32 head channels. All addressing/weights
// are wave-uniform -> scalar branches. Value is bf16 -> each corner is one
// coalesced 64B wave load from the L2-resident head-major value tensor
// (half the L2 traffic of f32); accumulation in f32.
// ---------------------------------------------------------------------------
__global__ __launch_bounds__(256) void msda_sample(
    const float*          __restrict__ refp,   // [N, Lq, 4, 2]
    const float*          __restrict__ offs,   // [N, Lq, H, L, P, 2]
    const float*          __restrict__ attn,   // [N, Lq, H, L*P]
    const unsigned short* __restrict__ value,  // [N, H, LENIN, 32] bf16
    float*                __restrict__ outbuf) // [N, Lq, H*32]
{
    const int g    = blockIdx.x * 8 + (threadIdx.x >> 5);  // (n*Lq+q)*H + h
    const int lane = threadIdx.x & 31;
    const int h    = g & (NHEADS - 1);
    const int nq   = g >> 3;
    const int n    = nq >> 11;                             // Lq = 2048

    const int Hs[NLEVELS] = {92, 46, 23, 12};
    const int Wz[NLEVELS] = {160, 80, 40, 20};
    const int St[NLEVELS] = {0, 14720, 18400, 19320};

    const float* rp = refp + (size_t)nq * (NLEVELS * 2);
    const float* op = offs + (size_t)g * (NLEVELS * NPOINTS * 2);
    const float* ap = attn + (size_t)g * (NLEVELS * NPOINTS);
    const unsigned short* vbase = value + (size_t)(n * NHEADS + h) * LENIN * DHEAD;

    float acc = 0.f;
    #pragma unroll
    for (int l = 0; l < NLEVELS; ++l) {
        const int   Hl = Hs[l], Wl = Wz[l];
        const float rx = rp[l * 2 + 0];
        const float ry = rp[l * 2 + 1];
        const unsigned short* vl = vbase + (size_t)St[l] * DHEAD;

        for (int p = 0; p < NPOINTS; ++p) {
            const float ox = op[(l * NPOINTS + p) * 2 + 0];
            const float oy = op[(l * NPOINTS + p) * 2 + 1];
            const float aw = ap[l * NPOINTS + p];

            // loc = ref + off/normalizer; x = loc_x*W - 0.5 = ref_x*W + off_x - 0.5
            const float x = rx * (float)Wl + ox - 0.5f;
            const float y = ry * (float)Hl + oy - 0.5f;
            const float x0f = floorf(x), y0f = floorf(y);
            const float fx = x - x0f, fy = y - y0f;
            const int xi = (int)x0f, yi = (int)y0f;

            const float w00 = (1.f - fx) * (1.f - fy) * aw;
            const float w10 = fx * (1.f - fy) * aw;
            const float w01 = (1.f - fx) * fy * aw;
            const float w11 = fx * fy * aw;

            const bool vx0 = (xi >= 0) && (xi < Wl);
            const bool vx1 = (xi + 1 >= 0) && (xi + 1 < Wl);
            if ((yi >= 0) && (yi < Hl)) {
                const unsigned short* r0 = vl + (size_t)(yi * Wl) * DHEAD;
                if (vx0) acc += w00 * bf2f(r0[(size_t)xi * DHEAD + lane]);
                if (vx1) acc += w10 * bf2f(r0[(size_t)(xi + 1) * DHEAD + lane]);
            }
            if ((yi + 1 >= 0) && (yi + 1 < Hl)) {
                const unsigned short* r1 = vl + (size_t)((yi + 1) * Wl) * DHEAD;
                if (vx0) acc += w01 * bf2f(r1[(size_t)xi * DHEAD + lane]);
                if (vx1) acc += w11 * bf2f(r1[(size_t)(xi + 1) * DHEAD + lane]);
            }
        }
    }
    outbuf[(size_t)nq * D_MODEL + h * DHEAD + lane] = acc;
}

// ---------------------------------------------------------------------------
extern "C" void kernel_launch(void* const* d_in, const int* in_sizes, int n_in,
                              void* d_out, int out_size, void* d_ws, size_t ws_size,
                              hipStream_t stream)
{
    const float* query  = (const float*)d_in[0];
    const float* refp   = (const float*)d_in[1];
    const float* inpf   = (const float*)d_in[2];
    // d_in[3] spatial shapes / d_in[4] level starts: baked as constants
    const float* W_samp = (const float*)d_in[5];
    const float* b_samp = (const float*)d_in[6];
    const float* W_attn = (const float*)d_in[7];
    const float* b_attn = (const float*)d_in[8];
    const float* W_v    = (const float*)d_in[9];
    const float* b_v    = (const float*)d_in[10];
    const float* W_out  = (const float*)d_in[11];
    const float* b_out  = (const float*)d_in[12];
    float* out = (float*)d_out;

    const int MQ   = NBATCH * LQ;        // 8192
    const int MV   = NBATCH * LENIN;     // 78240
    const int ROWS = MQ * NHEADS;        // 65536 (n,q,h) rows

    char* ws = (char*)d_ws;
    const size_t sz_val = (size_t)MV * D_MODEL * 2;                    // 40.1 MB (bf16)
    const size_t sz_off = (size_t)MQ * (NHEADS*NLEVELS*NPOINTS*2) * 4; // 41.9 MB
    const size_t sz_att = (size_t)MQ * (NHEADS*NLEVELS*NPOINTS) * 4;   // 21.0 MB
    unsigned short* v_val = (unsigned short*)(ws);
    float* v_off  = (float*)(ws + sz_val);
    float* v_att  = (float*)(ws + sz_val + sz_off);
    float* v_obuf = (float*)(ws + sz_val + sz_off + sz_att);           // 8.4 MB

    const dim3 blk(64);

    // 1) value = input_flatten @ W_v + b_v -> bf16, scattered to [n][h][pos][32]
    gemm_bf16_wmma<<<dim3(MV / 32, D_MODEL / 128), blk, 0, stream>>>(
        inpf, W_v, b_v, v_val, MV, D_MODEL, 1);

    // 2) sampling offsets = query @ W_samp + b_samp   [8192 x 1280]
    gemm_bf16_wmma<<<dim3(MQ / 32, 1280 / 128), blk, 0, stream>>>(
        query, W_samp, b_samp, v_off, MQ, 1280, 0);

    // 3) logits = query @ W_attn + b_attn             [8192 x 640]
    gemm_bf16_wmma<<<dim3(MQ / 32, 640 / 128), blk, 0, stream>>>(
        query, W_attn, b_attn, v_att, MQ, 640, 0);

    // 4) softmax over the 80 (level,point) logits per (n,q,h), in place
    softmax80<<<ROWS / 4, 128, 0, stream>>>(v_att, ROWS);

    // 5) bilinear sampling + weighted accumulation -> [N, Lq, 256]
    msda_sample<<<ROWS / 8, 256, 0, stream>>>(refp, v_off, v_att, v_val, v_obuf);

    // 6) out = obuf @ W_out + b_out
    gemm_bf16_wmma<<<dim3(MQ / 32, D_MODEL / 128), blk, 0, stream>>>(
        v_obuf, W_out, b_out, out, MQ, D_MODEL, 0);
}
